// BootstrapRecurrentMixer_18640158065262
// MI455X (gfx1250) — matrix operand order repaired and checked by reading, not compiled, not run on hardware
//
#include <hip/hip_runtime.h>
#include <hip/hip_bf16.h>

// ---------------------------------------------------------------------------
// Problem constants (match reference)
// ---------------------------------------------------------------------------
constexpr int Bb    = 4;
constexpr int S     = 2048;
constexpr int D     = 1024;
constexpr int INNER = 2048;
constexpr int STATE = 512;
constexpr int MS    = Bb * S;          // 8192 rows for the big GEMMs
constexpr int NBLK  = 16;              // persistent scan workgroups
constexpr int PHALF = NBLK * Bb * STATE;  // one partial buffer (floats)

// LDS row pads: stride % 64 dwords == 4 -> 16-row b128 reads hit all 64 banks
constexpr int LDS_STATE_LD = STATE + 8;   // 520 elements (1040 B / row)
constexpr int LDS_MIX_LD   = 128 + 8;     // 136 elements (272 B / row)

typedef __attribute__((ext_vector_type(16))) __bf16 v16bf;
typedef __attribute__((ext_vector_type(8)))  __bf16 v8bf;
typedef __attribute__((ext_vector_type(8)))  float  v8f;
typedef __attribute__((ext_vector_type(4)))  int    v4i_t;

// ---------------------------------------------------------------------------
// CDNA5 async global->LDS copy (guarded; falls back to plain copy)
// Builtin prototype (probe-verified): (int4 AS1*, int4 AS3*, Ii, Ii)
// ---------------------------------------------------------------------------
#if __has_builtin(__builtin_amdgcn_global_load_async_to_lds_b128)
#define HAVE_ASYNC_LDS 1
#endif

__device__ inline void copy16_g2l(__bf16* l, const __bf16* g) {
#ifdef HAVE_ASYNC_LDS
  __builtin_amdgcn_global_load_async_to_lds_b128(
      (__attribute__((address_space(1))) v4i_t*)const_cast<__bf16*>(g),
      (__attribute__((address_space(3))) v4i_t*)l, 0, 0);
#else
  *(v8bf*)l = *(const v8bf*)g;
#endif
}

__device__ inline void async_copy_wait() {
#ifdef HAVE_ASYNC_LDS
#if __has_builtin(__builtin_amdgcn_s_wait_asynccnt)
  __builtin_amdgcn_s_wait_asynccnt(0);
#else
  asm volatile("s_wait_asynccnt 0x0" ::: "memory");
#endif
#endif
}

// ---------------------------------------------------------------------------
// WMMA helpers (CDNA5: D = A[16x32] * B[32x16] + C, bf16 in / f32 accum)
// ---------------------------------------------------------------------------
__device__ inline v8f wmma_bf16(v16bf a, v16bf b, v8f c) {
  return __builtin_amdgcn_wmma_f32_16x16x32_bf16(
      /*neg_a=*/false, a, /*neg_b=*/false, b,
      /*c_mod=*/(short)0, c, /*reuse_a=*/false, /*reuse_b=*/false);
}

// A-matrix 16x32 (row-major source A[row][k], lda in elements).
// Code's k order (not ISA table order): lane L holds M=L%16; K = (L/16)*16 + 0..15 contiguous (matches B).
// Works for both global and LDS-derived generic pointers.
__device__ inline v16bf load_a16x32(const __bf16* A, int lda, int mbase,
                                    int kk, int lane) {
  int row = mbase + (lane & 15);
  int kb  = kk + ((lane >> 4) << 4);
  const __bf16* p = A + (size_t)row * lda + kb;
  v8bf lo = *(const v8bf*)(p);
  v8bf hi = *(const v8bf*)(p + 8);
  return __builtin_shufflevector(lo, hi, 0, 1, 2, 3, 4, 5, 6, 7,
                                 8, 9, 10, 11, 12, 13, 14, 15);
}

// B-matrix 32x16 where B[k][n] = W[n][k], W row-major [N][K] (ldw elements).
// Code's k order (not ISA table order): lane L holds N=L%16; K = (L/16)*16 + 0..15 contiguous.
__device__ inline v16bf load_b32x16(const __bf16* __restrict__ W, int ldw,
                                    int nbase, int kk, int lane) {
  int col = nbase + (lane & 15);
  int kb  = kk + ((lane >> 4) << 4);
  return *(const v16bf*)(W + (size_t)col * ldw + kb);
}

// ---------------------------------------------------------------------------
// fp32 -> bf16 conversion
// ---------------------------------------------------------------------------
__global__ void f32_to_bf16(const float* __restrict__ src,
                            __bf16* __restrict__ dst, size_t n) {
  size_t i = (size_t)blockIdx.x * blockDim.x + threadIdx.x;
  if (i < n) dst[i] = (__bf16)src[i];
}

// ---------------------------------------------------------------------------
// Generic bf16 WMMA GEMM: C[m,n] = sum_k A[m,k]*W[n,k] + bias[n]
// Block = 256 threads = 8 waves (4M x 2N); block tile 128 x 128;
// wave tile 32(M) x 64(N) = 2x4 accumulators. Software-pipelined k-loop.
// ---------------------------------------------------------------------------
template <bool OUT_BF16>
__launch_bounds__(256) __global__
void gemm_bf16(const __bf16* __restrict__ A, const __bf16* __restrict__ W,
               const float* __restrict__ bias, void* __restrict__ Cout,
               int M, int N, int K) {
  const int lane  = threadIdx.x & 31;
  const int w     = threadIdx.x >> 5;
  const int mbase = blockIdx.x * 128 + (w >> 1) * 32;
  const int nbase = blockIdx.y * 128 + (w & 1) * 64;

  v8f acc[2][4] = {};
  v16bf a0[2], b0[4], a1[2], b1[4];

  // prologue: load k=0 tiles
  #pragma unroll
  for (int mi = 0; mi < 2; ++mi) a0[mi] = load_a16x32(A, K, mbase + mi * 16, 0, lane);
  #pragma unroll
  for (int ni = 0; ni < 4; ++ni) b0[ni] = load_b32x16(W, K, nbase + ni * 16, 0, lane);

  // K is a multiple of 64 for every GEMM in this model.
  for (int kk = 0; kk < K; kk += 64) {
    // prefetch kk+32 (always in range), multiply kk
    #pragma unroll
    for (int mi = 0; mi < 2; ++mi) a1[mi] = load_a16x32(A, K, mbase + mi * 16, kk + 32, lane);
    #pragma unroll
    for (int ni = 0; ni < 4; ++ni) b1[ni] = load_b32x16(W, K, nbase + ni * 16, kk + 32, lane);
    #pragma unroll
    for (int mi = 0; mi < 2; ++mi)
      #pragma unroll
      for (int ni = 0; ni < 4; ++ni)
        acc[mi][ni] = wmma_bf16(a0[mi], b0[ni], acc[mi][ni]);

    // prefetch kk+64 (if any), multiply kk+32
    if (kk + 64 < K) {
      #pragma unroll
      for (int mi = 0; mi < 2; ++mi) a0[mi] = load_a16x32(A, K, mbase + mi * 16, kk + 64, lane);
      #pragma unroll
      for (int ni = 0; ni < 4; ++ni) b0[ni] = load_b32x16(W, K, nbase + ni * 16, kk + 64, lane);
    }
    #pragma unroll
    for (int mi = 0; mi < 2; ++mi)
      #pragma unroll
      for (int ni = 0; ni < 4; ++ni)
        acc[mi][ni] = wmma_bf16(a1[mi], b1[ni], acc[mi][ni]);
  }

  // D layout: lane = N (mod 16), VGPR r holds row M = r + (lane/16)*8.
  #pragma unroll
  for (int mi = 0; mi < 2; ++mi) {
    #pragma unroll
    for (int ni = 0; ni < 4; ++ni) {
      int col  = nbase + ni * 16 + (lane & 15);
      float bv = bias[col];
      #pragma unroll
      for (int r = 0; r < 8; ++r) {
        int row = mbase + mi * 16 + r + ((lane >> 4) << 3);
        float v = acc[mi][ni][r] + bv;
        if (OUT_BF16)
          ((__bf16*)Cout)[(size_t)row * N + col] = (__bf16)v;
        else
          ((float*)Cout)[(size_t)row * N + col] = v;
      }
    }
  }
}

// ---------------------------------------------------------------------------
// Scan init: state_bf [16 x STATE] (rows 0..3 = state0, rest 0), barrier = 0.
// ---------------------------------------------------------------------------
__global__ void scan_init(const float* __restrict__ state0,
                          __bf16* __restrict__ state_bf,
                          int* __restrict__ ctr) {
  int i = blockIdx.x * blockDim.x + threadIdx.x;
  if (i == 0) *ctr = 0;
  if (i < 16 * STATE) {
    float v = (i < Bb * STATE) ? state0[i] : 0.0f;
    state_bf[i] = (__bf16)v;
  }
}

// Grid-wide barrier for NBLK persistent blocks (monotone counter, no reset).
__device__ inline void gbar(int* ctr, int epoch) {
  __syncthreads();
  if (threadIdx.x == 0) {
    __threadfence();
    __hip_atomic_fetch_add(ctr, 1, __ATOMIC_RELEASE, __HIP_MEMORY_SCOPE_AGENT);
    while (__hip_atomic_load(ctr, __ATOMIC_ACQUIRE, __HIP_MEMORY_SCOPE_AGENT) <
           epoch * NBLK) {
      __builtin_amdgcn_s_sleep(8);
    }
    __threadfence();
  }
  __syncthreads();
}

// ---------------------------------------------------------------------------
// Persistent recurrent scan. 16 blocks x 256 threads (128 waves).
// ONE device-wide barrier per step:
//   P1: wave (blk*8+w) owns u-tile cu=(blk*8+w)*16 and matching g-tile of
//       sp = state @ Wsp^T (state read from block-local LDS; Wsp tiles live
//       in registers for the whole scan; A-tile ds_loads software-pipelined).
//       Elementwise tanh/sigmoid -> mixed to LDS (this block's 128-column
//       slice == its P2 K-slice) + global history for out_proj.
//   P2: new_state partials from LDS mixed slice -> partial[t&1] (global).
//   gbar: the only cross-block sync. Double-buffered partials make the
//         fast-block write / slow-block read race impossible.
//   reduce: EVERY block redundantly sums the 16 fp32 partials in identical
//       order (bitwise-identical result) + bias -> bf16 state rows 0..3 in
//       its own LDS. No global state round-trip, no second barrier.
// ---------------------------------------------------------------------------
__launch_bounds__(256, 1) __global__
void scan_kernel(const __bf16* __restrict__ proj,      // [B,S,3*INNER] bf16
                 const __bf16* __restrict__ Wsp,       // [2*INNER, STATE] bf16
                 const __bf16* __restrict__ Wsu,       // [STATE, INNER] bf16
                 const float* __restrict__ bsp,        // [2*INNER]
                 const float* __restrict__ bsu,        // [STATE]
                 const __bf16* __restrict__ state_bf,  // [16, STATE] initial
                 float* __restrict__ state_f32,        // [B, STATE] final out
                 __bf16* __restrict__ mixed_all,       // [B, S, INNER]
                 float* __restrict__ partial,          // [2][NBLK, B, STATE]
                 int* __restrict__ ctr) {
  __shared__ __bf16 lds_state[16 * LDS_STATE_LD];
  __shared__ __bf16 lds_mixed[16 * LDS_MIX_LD];

  const int tid    = threadIdx.x;
  const int lane   = tid & 31;
  const int w      = tid >> 5;
  const int blk    = blockIdx.x;
  const int wave_g = blk * 8 + w;      // 0..127
  const int cu     = wave_g * 16;      // u-column tile base (0..2047)
  int epoch = 0;

  // zero the padded mixed buffer once (rows 4..15 stay zero forever)
  for (int i = tid; i < 16 * LDS_MIX_LD; i += 256) lds_mixed[i] = (__bf16)0.0f;

  // initial state -> LDS via async DMA (rows 4..15 are zero in global)
  #pragma unroll
  for (int c = 0; c < 4; ++c) {
    int chunk = tid + c * 256;                 // 1024 chunks of 8 bf16
    int row   = chunk >> 6;
    int col8  = (chunk & 63) << 3;
    copy16_g2l(&lds_state[row * LDS_STATE_LD + col8],
               state_bf + row * STATE + col8);
  }
  async_copy_wait();

  // preload this wave's Wsp tiles (u and g) for the whole scan
  v16bf bU[16], bG[16];
  #pragma unroll
  for (int ks = 0; ks < 16; ++ks) {
    bU[ks] = load_b32x16(Wsp, STATE, cu, ks * 32, lane);
    bG[ks] = load_b32x16(Wsp, STATE, INNER + cu, ks * 32, lane);
  }
  __syncthreads();

  for (int t = 0; t < S; ++t) {
    float* part = partial + ((t & 1) ? PHALF : 0);

    // ---------------- P1: sp tiles + elementwise mix -----------------
    v8f accU = {};
    v8f accG = {};
    v16bf aa = load_a16x32(lds_state, LDS_STATE_LD, 0, 0, lane);
    #pragma unroll
    for (int ks = 0; ks < 16; ++ks) {
      v16bf cur = aa;
      if (ks < 15)   // prefetch next A tile while multiplying this one
        aa = load_a16x32(lds_state, LDS_STATE_LD, 0, (ks + 1) * 32, lane);
      accU = wmma_bf16(cur, bU[ks], accU);
      accG = wmma_bf16(cur, bG[ks], accG);
    }
    {
      int col     = cu + (lane & 15);
      int lcol    = w * 16 + (lane & 15);        // block-local mixed column
      float biasU = bsp[col];
      float biasG = bsp[INNER + col];
      #pragma unroll
      for (int r = 0; r < 8; ++r) {
        int b = r + ((lane >> 4) << 3);          // accumulator row = batch
        if (b < Bb) {
          size_t base = ((size_t)b * S + t) * (3 * INNER);
          float tu = (float)proj[base + col];
          float tg = (float)proj[base + INNER + col];
          float tv = (float)proj[base + 2 * INNER + col];
          float u    = accU[r] + biasU + tu;
          float g    = accG[r] + biasG + tg;
          float cand = tanhf(u);
          float gate = 1.0f / (1.0f + __expf(-g));
          float mx   = gate * cand + (1.0f - gate) * tanhf(tv);
          lds_mixed[b * LDS_MIX_LD + lcol] = (__bf16)mx;
          mixed_all[((size_t)b * S + t) * INNER + col] = (__bf16)mx;
        }
      }
    }
    __syncthreads();   // LDS-local handoff (P1 cols == this block's K slice)

    // ---------------- P2: new_state partials -------------------------
    {
      const int jbase = blk * 128;               // global K offset of slice
      v16bf am[4];                               // shared across 4 N-tiles
      #pragma unroll
      for (int kk = 0; kk < 4; ++kk)
        am[kk] = load_a16x32(lds_mixed, LDS_MIX_LD, 0, kk * 32, lane);
      #pragma unroll
      for (int i = 0; i < 4; ++i) {
        int nb = (w * 4 + i) * 16;               // s-column tile (0..511)
        v8f acc = {};
        #pragma unroll
        for (int kk = 0; kk < 4; ++kk) {
          v16bf bb = load_b32x16(Wsu, INNER, nb, jbase + kk * 32, lane);
          acc = wmma_bf16(am[kk], bb, acc);
        }
        int scol = nb + (lane & 15);
        #pragma unroll
        for (int r = 0; r < 8; ++r) {
          int b = r + ((lane >> 4) << 3);
          if (b < Bb)
            part[((size_t)blk * Bb + b) * STATE + scol] = acc[r];
        }
      }
    }
    gbar(ctr, ++epoch);   // the only device-wide sync this step

    // ------ reduce: every block redundantly -> its own LDS state ------
    {
      int i0 = tid * 8;                          // 8 contiguous s per thread
      int b  = i0 >> 9;
      int s  = i0 & (STATE - 1);
      float sum[8];
      #pragma unroll
      for (int c = 0; c < 8; ++c) sum[c] = bsu[s + c];
      #pragma unroll
      for (int gg = 0; gg < NBLK; ++gg) {
        const float* pr = part + ((size_t)gg * Bb + b) * STATE + s;
        #pragma unroll
        for (int c = 0; c < 8; ++c) sum[c] += pr[c];
      }
      #pragma unroll
      for (int c = 0; c < 8; ++c)
        lds_state[b * LDS_STATE_LD + s + c] = (__bf16)sum[c];
      if (t == S - 1 && blk == 0) {
        #pragma unroll
        for (int c = 0; c < 8; ++c) state_f32[i0 + c] = sum[c];
      }
    }
    __syncthreads();   // state visible block-wide before next P1
  }
}

__global__ void copy_final_state(const float* __restrict__ state_f32,
                                 float* __restrict__ out_tail) {
  int i = blockIdx.x * blockDim.x + threadIdx.x;
  if (i < Bb * STATE) out_tail[i] = state_f32[i];
}

// ---------------------------------------------------------------------------
// Launcher
// ---------------------------------------------------------------------------
extern "C" void kernel_launch(void* const* d_in, const int* in_sizes, int n_in,
                              void* d_out, int out_size, void* d_ws,
                              size_t ws_size, hipStream_t stream) {
  const float* hidden = (const float*)d_in[0];
  const float* state0 = (const float*)d_in[1];
  const float* w_in   = (const float*)d_in[2];
  const float* b_in   = (const float*)d_in[3];
  const float* w_sp   = (const float*)d_in[4];
  const float* b_sp   = (const float*)d_in[5];
  const float* w_out  = (const float*)d_in[6];
  const float* b_out  = (const float*)d_in[7];
  const float* w_su   = (const float*)d_in[8];
  const float* b_su   = (const float*)d_in[9];

  // Workspace carve (256B aligned slices)
  char* ws = (char*)d_ws;
  size_t off = 0;
  auto carve = [&](size_t bytes) -> void* {
    void* p = ws + off;
    off += (bytes + 255) & ~(size_t)255;
    return p;
  };
  __bf16* hid_bf    = (__bf16*)carve((size_t)MS * D * 2);
  __bf16* w_in_bf   = (__bf16*)carve((size_t)3 * INNER * D * 2);
  __bf16* w_sp_bf   = (__bf16*)carve((size_t)2 * INNER * STATE * 2);
  __bf16* w_su_bf   = (__bf16*)carve((size_t)STATE * INNER * 2);
  __bf16* w_out_bf  = (__bf16*)carve((size_t)D * INNER * 2);
  __bf16* proj_bf   = (__bf16*)carve((size_t)MS * 3 * INNER * 2);
  __bf16* mixed_bf  = (__bf16*)carve((size_t)MS * INNER * 2);
  __bf16* state_bf  = (__bf16*)carve((size_t)16 * STATE * 2);
  float*  state_f32 = (float*)carve((size_t)Bb * STATE * 4);
  float*  partial   = (float*)carve((size_t)2 * PHALF * 4);  // double-buffered
  int*    ctr       = (int*)carve(256);

  // 1) convert fp32 -> bf16 operands
  auto cvt = [&](const float* s, __bf16* d, size_t n) {
    f32_to_bf16<<<(unsigned)((n + 255) / 256), 256, 0, stream>>>(s, d, n);
  };
  cvt(hidden, hid_bf,  (size_t)MS * D);
  cvt(w_in,   w_in_bf, (size_t)3 * INNER * D);
  cvt(w_sp,   w_sp_bf, (size_t)2 * INNER * STATE);
  cvt(w_su,   w_su_bf, (size_t)STATE * INNER);
  cvt(w_out,  w_out_bf,(size_t)D * INNER);

  // 2) scan state init + barrier counter reset
  scan_init<<<(16 * STATE + 255) / 256, 256, 0, stream>>>(state0, state_bf, ctr);

  // 3) in_proj: projected = hidden @ in_proj_w^T + b  -> bf16 [B,S,3*INNER]
  gemm_bf16<true><<<dim3(MS / 128, (3 * INNER) / 128), 256, 0, stream>>>(
      hid_bf, w_in_bf, b_in, proj_bf, MS, 3 * INNER, D);

  // 4) recurrent scan (persistent, 1 grid sync per step)
  scan_kernel<<<NBLK, 256, 0, stream>>>(proj_bf, w_sp_bf, w_su_bf, b_sp, b_su,
                                        state_bf, state_f32, mixed_bf,
                                        partial, ctr);

  // 5) out_proj: out = mixed @ out_proj_w^T + b -> fp32 [B,S,D]
  gemm_bf16<false><<<dim3(MS / 128, D / 128), 256, 0, stream>>>(
      mixed_bf, w_out_bf, b_out, d_out, MS, D, INNER);

  // 6) final_state tail of d_out
  copy_final_state<<<(Bb * STATE + 255) / 256, 256, 0, stream>>>(
      state_f32, (float*)d_out + (size_t)MS * D);
}
